// MultiHeadAttention_58445914964307
// MI455X (gfx1250) — matrix operand order repaired and checked by reading, not compiled, not run on hardware
//
#include <hip/hip_runtime.h>
#include <hip/hip_bf16.h>

// ---------------------------------------------------------------------------
// MI455X (gfx1250, wave32) fused multi-head attention.
// All matmuls via v_wmma_f32_16x16x32_f16 (f32 accumulate, f16 operands).
// Round 2: low register pressure (3-pass QKV), no unions (pure ext-vector
// SSA values), double-buffered LDS weight panels (1 barrier per K-step).
// ---------------------------------------------------------------------------

typedef __attribute__((ext_vector_type(16))) _Float16 v16h;
typedef __attribute__((ext_vector_type(8)))  _Float16 v8h;
typedef __attribute__((ext_vector_type(4)))  _Float16 v4h;
typedef __attribute__((ext_vector_type(8)))  float    v8f;

#define NB   1024   // batch
#define NT   64     // seq len
#define NC   768    // channels
#define NH   12     // heads
#define ND   64     // head dim
#define NKP  24     // K-panels of 32 over C=768

#define ZERO8F {0.f,0.f,0.f,0.f,0.f,0.f,0.f,0.f}

static __device__ __forceinline__ v16h cat16(v8h lo, v8h hi) {
    return __builtin_shufflevector(lo, hi, 0,1,2,3,4,5,6,7,8,9,10,11,12,13,14,15);
}

// A/B fragment from two contiguous 16B runs
static __device__ __forceinline__ v16h ld_frag(const _Float16* p0, const _Float16* p1) {
    return cat16(*(const v8h*)p0, *(const v8h*)p1);
}

static __device__ __forceinline__ v8f wmma_f16(v16h a, v16h b, v8f c) {
    return __builtin_amdgcn_wmma_f32_16x16x32_f16(false, a, false, b,
                                                  (short)0, c, false, false);
}

// stage one 2048-half (4KB) contiguous panel with 128 threads
static __device__ __forceinline__ void stage_panel(_Float16* dst,
                                                   const _Float16* src, int tid) {
    const int idx = tid * 16;
    *(v8h*)(dst + idx)     = *(const v8h*)(src + idx);
    *(v8h*)(dst + idx + 8) = *(const v8h*)(src + idx + 8);
}

// ---------------------------------------------------------------------------
// Pack kernels: f32 -> f16 + WMMA-friendly panel layouts (bandwidth trivial)
// ---------------------------------------------------------------------------

// x [B,T,C] f32 -> Xh [B*T, C] f16
__global__ __launch_bounds__(256) void pack_x_kernel(const float* __restrict__ x,
                                                     _Float16* __restrict__ Xh) {
    size_t i = (size_t)blockIdx.x * 256 + threadIdx.x;   // one float4 each
    float4 v = ((const float4*)x)[i];
    v4h o;
    o[0] = (_Float16)v.x; o[1] = (_Float16)v.y;
    o[2] = (_Float16)v.z; o[3] = (_Float16)v.w;
    ((v4h*)Xh)[i] = o;
}

// Wq/Wk/Wv [H,D,C] f32 -> WP [h][kp][d][ki] f16  (panel-major for B frags)
__global__ __launch_bounds__(256) void pack_wqkv_kernel(
    const float* __restrict__ Wq, const float* __restrict__ Wk,
    const float* __restrict__ Wv,
    _Float16* __restrict__ WqP, _Float16* __restrict__ WkP,
    _Float16* __restrict__ WvP) {
    int tid = blockIdx.x * 256 + threadIdx.x;
    const int n = NH * NKP * ND * 32;   // 589824 per matrix
    int mat = tid / n;
    int rem = tid - mat * n;
    const float* src = (mat == 0) ? Wq : ((mat == 1) ? Wk : Wv);
    _Float16*    dst = (mat == 0) ? WqP : ((mat == 1) ? WkP : WvP);
    int h  = rem / (NKP * ND * 32);            // / 49152
    int r2 = rem - h * (NKP * ND * 32);
    int kp = r2 >> 11;                          // / 2048
    int r3 = r2 & 2047;
    int d  = r3 >> 5;
    int ki = r3 & 31;
    dst[rem] = (_Float16)src[(h * ND + d) * NC + kp * 32 + ki];
}

// Wp [C, H*D] f32 -> WpP [kp][c][ki] f16  (B[k=j][n=c] = Wp[c][j])
__global__ __launch_bounds__(256) void pack_wp_kernel(const float* __restrict__ Wp,
                                                      _Float16* __restrict__ WpP) {
    int tid = blockIdx.x * 256 + threadIdx.x;   // exact: 24*768*32
    int ki = tid & 31;
    int c  = (tid >> 5) % NC;
    int kp = tid / (NC * 32);
    WpP[tid] = (_Float16)Wp[c * NC + kp * 32 + ki];
}

// ---------------------------------------------------------------------------
// Fused attention: one block per (b, h). 128 threads = 4 wave32.
// Wave w owns token rows [16w, 16w+16).
// ---------------------------------------------------------------------------
__global__ __launch_bounds__(128) void attn_kernel(
    const _Float16* __restrict__ Xh,
    const _Float16* __restrict__ WqP, const _Float16* __restrict__ WkP,
    const _Float16* __restrict__ WvP,
    _Float16* __restrict__ Attn) {

    const int b    = blockIdx.x;
    const int h    = blockIdx.y;
    const int tid  = threadIdx.x;
    const int lane = tid & 31;
    const int w    = tid >> 5;      // wave 0..3
    const int r    = lane & 15;     // N-lane / row-lane
    const int hf   = lane >> 4;     // half of wave

    __shared__ _Float16 sW[2 * 2048];     // double-buffered weight panel  8KB
    __shared__ _Float16 sQ[64 * 64];      // Q (scaled)                    8KB
    __shared__ _Float16 sK[64 * 64];      // K row-major [s][d]            8KB
    __shared__ _Float16 sVt[64 * 64];     // V transposed [d][s]           8KB
    __shared__ _Float16 sP[64 * 64];      // softmax probs                 8KB

    const _Float16* xrow = Xh + ((size_t)b * NT + 16 * w + r) * NC;
    const _Float16* wsrc0 = WqP + (size_t)h * (NKP * 2048);
    const _Float16* wsrc1 = WkP + (size_t)h * (NKP * 2048);
    const _Float16* wsrc2 = WvP + (size_t)h * (NKP * 2048);

    // ---- QKV projection, one matrix at a time (low register pressure) ----
    for (int mat = 0; mat < 3; ++mat) {
        const _Float16* wsrc = (mat == 0) ? wsrc0 : (mat == 1) ? wsrc1 : wsrc2;

        v8f acc0 = ZERO8F, acc1 = ZERO8F, acc2 = ZERO8F, acc3 = ZERO8F;

        stage_panel(sW, wsrc, tid);            // prologue: panel 0 -> buf 0
        __syncthreads();

        for (int kp = 0; kp < NKP; ++kp) {
            const int cur = kp & 1;
            if (kp + 1 < NKP)                  // overlap: stage next panel
                stage_panel(sW + (cur ^ 1) * 2048, wsrc + (kp + 1) * 2048, tid);

            v16h a = ld_frag(xrow + kp * 32 + hf * 16,
                             xrow + kp * 32 + hf * 16 + 8);
            if (kp + 1 < NKP) __builtin_prefetch(xrow + (kp + 1) * 32, 0, 3);

            const _Float16* wp = sW + cur * 2048 + r * 32 + hf * 16;
            v16h b0 = ld_frag(wp + 0 * 512, wp + 0 * 512 + 8);
            v16h b1 = ld_frag(wp + 1 * 512, wp + 1 * 512 + 8);
            v16h b2 = ld_frag(wp + 2 * 512, wp + 2 * 512 + 8);
            v16h b3 = ld_frag(wp + 3 * 512, wp + 3 * 512 + 8);
            acc0 = wmma_f16(a, b0, acc0);
            acc1 = wmma_f16(a, b1, acc1);
            acc2 = wmma_f16(a, b2, acc2);
            acc3 = wmma_f16(a, b3, acc3);
            __syncthreads();                   // single barrier per K-step
        }

        // write this matrix's 16x64 strip to LDS
        #pragma unroll
        for (int i = 0; i < 8; ++i) {
            const int row = 16 * w + hf * 8 + i;
            if (mat == 0) {
                sQ[row * 64 +  0 + r] = (_Float16)(acc0[i] * 0.125f);
                sQ[row * 64 + 16 + r] = (_Float16)(acc1[i] * 0.125f);
                sQ[row * 64 + 32 + r] = (_Float16)(acc2[i] * 0.125f);
                sQ[row * 64 + 48 + r] = (_Float16)(acc3[i] * 0.125f);
            } else if (mat == 1) {
                sK[row * 64 +  0 + r] = (_Float16)acc0[i];
                sK[row * 64 + 16 + r] = (_Float16)acc1[i];
                sK[row * 64 + 32 + r] = (_Float16)acc2[i];
                sK[row * 64 + 48 + r] = (_Float16)acc3[i];
            } else {
                sVt[( 0 + r) * 64 + row] = (_Float16)acc0[i];
                sVt[(16 + r) * 64 + row] = (_Float16)acc1[i];
                sVt[(32 + r) * 64 + row] = (_Float16)acc2[i];
                sVt[(48 + r) * 64 + row] = (_Float16)acc3[i];
            }
        }
    }
    __syncthreads();

    // ---- S = Q * K^T  (wave computes its 16 rows x all 64 cols) ----
    v8f s0 = ZERO8F, s1 = ZERO8F, s2 = ZERO8F, s3 = ZERO8F;
    #pragma unroll
    for (int ks = 0; ks < 2; ++ks) {
        const int qo = (16 * w + r) * 64 + ks * 32;
        v16h a = ld_frag(&sQ[qo + hf * 16], &sQ[qo + hf * 16 + 8]);
        const _Float16* kpp = sK + r * 64 + ks * 32 + hf * 16;   // B[k=d][n=s]
        s0 = wmma_f16(a, ld_frag(kpp +  0 * 1024, kpp +  0 * 1024 + 8), s0);
        s1 = wmma_f16(a, ld_frag(kpp +  1 * 1024, kpp +  1 * 1024 + 8), s1);
        s2 = wmma_f16(a, ld_frag(kpp +  2 * 1024, kpp +  2 * 1024 + 8), s2);
        s3 = wmma_f16(a, ld_frag(kpp +  3 * 1024, kpp +  3 * 1024 + 8), s3);
    }

    // ---- causal softmax (row-wise; 16-lane shuffle reduction per half) ----
    #pragma unroll
    for (int i = 0; i < 8; ++i) {
        const int row = 16 * w + hf * 8 + i;
        float v0 = ( 0 + r <= row) ? s0[i] : -3.0e38f;
        float v1 = (16 + r <= row) ? s1[i] : -3.0e38f;
        float v2 = (32 + r <= row) ? s2[i] : -3.0e38f;
        float v3 = (48 + r <= row) ? s3[i] : -3.0e38f;
        float m = fmaxf(fmaxf(v0, v1), fmaxf(v2, v3));
        m = fmaxf(m, __shfl_xor(m, 1, 32));
        m = fmaxf(m, __shfl_xor(m, 2, 32));
        m = fmaxf(m, __shfl_xor(m, 4, 32));
        m = fmaxf(m, __shfl_xor(m, 8, 32));
        float e0 = ( 0 + r <= row) ? __expf(v0 - m) : 0.f;
        float e1 = (16 + r <= row) ? __expf(v1 - m) : 0.f;
        float e2 = (32 + r <= row) ? __expf(v2 - m) : 0.f;
        float e3 = (48 + r <= row) ? __expf(v3 - m) : 0.f;
        float s = e0 + e1 + e2 + e3;
        s += __shfl_xor(s, 1, 32);
        s += __shfl_xor(s, 2, 32);
        s += __shfl_xor(s, 4, 32);
        s += __shfl_xor(s, 8, 32);
        float inv = 1.0f / s;
        sP[row * 64 +  0 + r] = (_Float16)(e0 * inv);
        sP[row * 64 + 16 + r] = (_Float16)(e1 * inv);
        sP[row * 64 + 32 + r] = (_Float16)(e2 * inv);
        sP[row * 64 + 48 + r] = (_Float16)(e3 * inv);
    }
    __syncthreads();

    // ---- O = P * V ----
    v8f o0 = ZERO8F, o1 = ZERO8F, o2 = ZERO8F, o3 = ZERO8F;
    #pragma unroll
    for (int ks = 0; ks < 2; ++ks) {
        const int po = (16 * w + r) * 64 + ks * 32;
        v16h a = ld_frag(&sP[po + hf * 16], &sP[po + hf * 16 + 8]);
        const _Float16* vpp = sVt + r * 64 + ks * 32 + hf * 16;  // B[k=s][n=d]
        o0 = wmma_f16(a, ld_frag(vpp + 0 * 1024, vpp + 0 * 1024 + 8), o0);
        o1 = wmma_f16(a, ld_frag(vpp + 1 * 1024, vpp + 1 * 1024 + 8), o1);
        o2 = wmma_f16(a, ld_frag(vpp + 2 * 1024, vpp + 2 * 1024 + 8), o2);
        o3 = wmma_f16(a, ld_frag(vpp + 3 * 1024, vpp + 3 * 1024 + 8), o3);
    }

    // ---- write head-concat output (f16) ----
    _Float16* outp = Attn + (size_t)b * NT * NC + (size_t)h * ND;
    #pragma unroll
    for (int i = 0; i < 8; ++i) {
        const int row = 16 * w + hf * 8 + i;
        outp[(size_t)row * NC +  0 + r] = (_Float16)o0[i];
        outp[(size_t)row * NC + 16 + r] = (_Float16)o1[i];
        outp[(size_t)row * NC + 32 + r] = (_Float16)o2[i];
        outp[(size_t)row * NC + 48 + r] = (_Float16)o3[i];
    }
}

// ---------------------------------------------------------------------------
// Output projection: out[65536,768] = Attn[65536,768] * Wp^T + bp
// Block = 64x64 output tile, 4 waves, double-buffered 4KB panel.
// ---------------------------------------------------------------------------
__global__ __launch_bounds__(128) void proj_kernel(
    const _Float16* __restrict__ Attn, const _Float16* __restrict__ WpP,
    const float* __restrict__ bp, float* __restrict__ out) {

    const int mblk = blockIdx.x;    // 64 token rows
    const int nblk = blockIdx.y;    // 64 output cols
    const int tid  = threadIdx.x;
    const int lane = tid & 31;
    const int w    = tid >> 5;
    const int r    = lane & 15;
    const int hf   = lane >> 4;

    __shared__ _Float16 sW[2 * 2048];   // double-buffered 4KB panel

    v8f acc0 = ZERO8F, acc1 = ZERO8F, acc2 = ZERO8F, acc3 = ZERO8F;

    const _Float16* arow = Attn + ((size_t)mblk * 64 + 16 * w + r) * NC;
    const _Float16* wbase = WpP + (size_t)nblk * 64 * 32;   // within each kp slab

    stage_panel(sW, wbase, tid);                  // kp = 0 -> buf 0
    __syncthreads();

    for (int kp = 0; kp < NKP; ++kp) {
        const int cur = kp & 1;
        if (kp + 1 < NKP)
            stage_panel(sW + (cur ^ 1) * 2048,
                        wbase + (size_t)(kp + 1) * NC * 32, tid);

        v16h a = ld_frag(arow + kp * 32 + hf * 16,
                         arow + kp * 32 + hf * 16 + 8);
        if (kp + 1 < NKP) __builtin_prefetch(arow + (kp + 1) * 32, 0, 3);

        const _Float16* wp = sW + cur * 2048 + r * 32 + hf * 16;
        acc0 = wmma_f16(a, ld_frag(wp + 0 * 512, wp + 0 * 512 + 8), acc0);
        acc1 = wmma_f16(a, ld_frag(wp + 1 * 512, wp + 1 * 512 + 8), acc1);
        acc2 = wmma_f16(a, ld_frag(wp + 2 * 512, wp + 2 * 512 + 8), acc2);
        acc3 = wmma_f16(a, ld_frag(wp + 3 * 512, wp + 3 * 512 + 8), acc3);
        __syncthreads();
    }

    const int colb = nblk * 64 + r;
    float bias0 = bp[colb +  0];
    float bias1 = bp[colb + 16];
    float bias2 = bp[colb + 32];
    float bias3 = bp[colb + 48];
    #pragma unroll
    for (int i = 0; i < 8; ++i) {
        const size_t row = (size_t)mblk * 64 + 16 * w + hf * 8 + i;
        out[row * NC + colb +  0] = acc0[i] + bias0;
        out[row * NC + colb + 16] = acc1[i] + bias1;
        out[row * NC + colb + 32] = acc2[i] + bias2;
        out[row * NC + colb + 48] = acc3[i] + bias3;
    }
}

// ---------------------------------------------------------------------------
// Launcher
// ---------------------------------------------------------------------------
extern "C" void kernel_launch(void* const* d_in, const int* in_sizes, int n_in,
                              void* d_out, int out_size, void* d_ws, size_t ws_size,
                              hipStream_t stream) {
    (void)in_sizes; (void)n_in; (void)out_size; (void)ws_size;

    const float* x  = (const float*)d_in[0];
    const float* Wq = (const float*)d_in[1];
    const float* Wk = (const float*)d_in[2];
    const float* Wv = (const float*)d_in[3];
    const float* Wp = (const float*)d_in[4];
    const float* bp = (const float*)d_in[5];
    float* out = (float*)d_out;

    // workspace layout (f16 scratch), ~197 MB total
    char* ws = (char*)d_ws;
    const size_t szXh = (size_t)NB * NT * NC * 2;          // 100663296
    const size_t szW  = (size_t)NH * NKP * ND * 32 * 2;    //   1179648
    const size_t szWp = (size_t)NKP * NC * 32 * 2;         //   1179648
    _Float16* Xh   = (_Float16*)(ws);
    _Float16* WqP  = (_Float16*)(ws + szXh);
    _Float16* WkP  = (_Float16*)(ws + szXh + szW);
    _Float16* WvP  = (_Float16*)(ws + szXh + 2 * szW);
    _Float16* WpP  = (_Float16*)(ws + szXh + 3 * szW);
    _Float16* Attn = (_Float16*)(ws + szXh + 3 * szW + szWp);

    // 1) pack inputs to f16 / WMMA panel layouts
    pack_x_kernel<<<(NB * NT * NC) / (256 * 4), 256, 0, stream>>>(x, Xh);
    pack_wqkv_kernel<<<(3 * NH * NKP * ND * 32) / 256, 256, 0, stream>>>(
        Wq, Wk, Wv, WqP, WkP, WvP);
    pack_wp_kernel<<<(NKP * NC * 32) / 256, 256, 0, stream>>>(Wp, WpP);

    // 2) fused QKV + causal attention per (batch, head)
    attn_kernel<<<dim3(NB, NH), 128, 0, stream>>>(Xh, WqP, WkP, WvP, Attn);

    // 3) output projection + bias
    proj_kernel<<<dim3(NB, NH), 128, 0, stream>>>(Attn, WpP, bp, out);
}
